// EdgeConvEncoder_12618613916263
// MI455X (gfx1250) — matrix operand order repaired and checked by reading, not compiled, not run on hardware
//
#include <hip/hip_runtime.h>
#include <hip/hip_bf16.h>

typedef __attribute__((ext_vector_type(16))) _Float16 v16h;
typedef __attribute__((ext_vector_type(8)))  _Float16 v8h;
typedef __attribute__((ext_vector_type(8)))  float    v8f;

#define EPS 1e-5f

// ---------------------------------------------------------------------------
// WMMA helpers (CDNA5 gfx1250: V_WMMA_F32_16X16X32_F16, wave32)
// ---------------------------------------------------------------------------
__device__ __forceinline__ v8f wmma32(v16h a, v16h b, v8f c) {
    // D = A(16x32 f16) * B(32x16 f16) + C(16x16 f32)
    return __builtin_amdgcn_wmma_f32_16x16x32_f16(
        /*neg_a=*/false, a, /*neg_b=*/false, b,
        /*c_mod=*/(short)0, c, /*reuse_a=*/false, /*reuse_b=*/false);
}

// A-matrix 16x32 f16 layout (ISA 7.12.2):
//   lane<16:  M=lane,    halves 0..7 -> K = kbase+8*kq+0..7 (kq=0)
//             halves 8..15 -> K = kbase+16+8*kq+0..7
//   lane>=16: M=lane-16, same with kq=1
// Both runs are 8 contiguous halves = one 16B load each.
__device__ __forceinline__ v16h load_a_row(const _Float16* row, int kbase, int kq) {
    const v8h lo = *(const v8h*)(row + kbase + 8 * kq);
    const v8h hi = *(const v8h*)(row + kbase + 16 + 8 * kq);
    v16h a;
#pragma unroll
    for (int i = 0; i < 8; ++i) { a[i] = lo[i]; a[i + 8] = hi[i]; }
    return a;
}

// A fragment built from f32 data with a per-row scale (for segment means)
__device__ __forceinline__ v16h load_a_f32_scaled(const float* row, int kbase, int kq, float inv) {
    v16h a;
#pragma unroll
    for (int i = 0; i < 8; ++i) {
        a[i]     = (_Float16)(row[kbase + 8 * kq + i] * inv);
        a[i + 8] = (_Float16)(row[kbase + 16 + 8 * kq + i] * inv);
    }
    return a;
}

// B-matrix 32x16 f16 layout: lane<16 -> N=lane, K=kbase+0..7,16..23;
// lane>=16 -> N=lane-16, K=kbase+8..15,24..31.  Weights are stored pre-transposed
// as WT[N][K] (K contiguous) so this is two 16B loads.
__device__ __forceinline__ v16h load_b_frag(const _Float16* wt, int kstride, int nbase,
                                            int kbase, int lane) {
    const _Float16* p = wt + (size_t)(nbase + (lane & 15)) * kstride + kbase + 8 * (lane >> 4);
    const v8h lo = *(const v8h*)p;
    const v8h hi = *(const v8h*)(p + 16);
    v16h b;
#pragma unroll
    for (int i = 0; i < 8; ++i) { b[i] = lo[i]; b[i + 8] = hi[i]; }
    return b;
}

__device__ __forceinline__ v8f splat8(float x) {
    v8f c;
#pragma unroll
    for (int i = 0; i < 8; ++i) c[i] = x;
    return c;
}

__device__ __forceinline__ void atomAddF(float* p, float v) { unsafeAtomicAdd(p, v); }

// wave-private LDS producer->consumer ordering (no cross-wave sharing)
__device__ __forceinline__ void wave_ds_fence() {
    asm volatile("s_wait_dscnt 0" ::: "memory");
}

// ---------------------------------------------------------------------------
// Utility kernels
// ---------------------------------------------------------------------------
__global__ __launch_bounds__(256) void zero_f32_kernel(float* p, long n) {
    long i = (long)blockIdx.x * blockDim.x + threadIdx.x;
    long stride = (long)gridDim.x * blockDim.x;
    for (; i < n; i += stride) p[i] = 0.0f;
}

// Transpose + f16-convert all weight matrices once. Row-major W[K][64] -> WT[64][K].
__global__ __launch_bounds__(256) void weight_prep_kernel(
    const float* __restrict__ W_init, const float* __restrict__ W_e1,
    const float* __restrict__ W_e2,  const float* __restrict__ W_u,
    const float* __restrict__ W_out,
    _Float16* __restrict__ wt_init, _Float16* __restrict__ wt_e1,
    _Float16* __restrict__ wt_e2,  _Float16* __restrict__ wt_u,
    _Float16* __restrict__ wt_out) {
    const int tid = threadIdx.x;
    for (int i = tid; i < 64 * 32; i += 256) { int n = i / 32, k = i % 32; wt_init[n * 32 + k] = (_Float16)W_init[k * 64 + n]; }
    for (int i = tid; i < 64 * 96; i += 256) { int n = i / 96, k = i % 96; wt_e1[n * 96 + k]  = (_Float16)W_e1[k * 64 + n]; }
    for (int i = tid; i < 64 * 64; i += 256) { int n = i / 64, k = i % 64; wt_e2[n * 64 + k]  = (_Float16)W_e2[k * 64 + n]; }
    for (int i = tid; i < 64 * 64; i += 256) { int n = i / 64, k = i % 64; wt_u[n * 64 + k]   = (_Float16)W_u[k * 64 + n]; }
    for (int i = tid; i < 64 * 64; i += 256) { int n = i / 64, k = i % 64; wt_out[n * 64 + k] = (_Float16)W_out[k * 64 + n]; }
}

// ---------------------------------------------------------------------------
// BatchNorm: partial sums (stats[0..31]=sum, stats[32..63]=sumsq)
// ---------------------------------------------------------------------------
__global__ __launch_bounds__(256) void bn_partial_kernel(
    const float* __restrict__ edge_attr, float* __restrict__ stats, int E, int chunk) {
    __shared__ float ssum[256], ssq[256];
    const int tid = threadIdx.x;
    const int f = tid & 31, g = tid >> 5;
    const int e0 = blockIdx.x * chunk;
    const int e1 = min(E, e0 + chunk);
    float s = 0.f, q = 0.f;
    for (int e = e0 + g; e < e1; e += 8) {
        float v = edge_attr[(size_t)e * 32 + f];
        s += v; q += v * v;
    }
    ssum[tid] = s; ssq[tid] = q;
    __syncthreads();
    if (tid < 32) {
#pragma unroll
        for (int gg = 1; gg < 8; ++gg) { s += ssum[gg * 32 + f]; q += ssq[gg * 32 + f]; }
        atomAddF(&stats[f], s);
        atomAddF(&stats[32 + f], q);
    }
}

// stats -> scale/shift:  ea = a*scale[f] + shift[f]
__global__ void bn_finalize_kernel(const float* __restrict__ gamma, const float* __restrict__ beta,
                                   float* __restrict__ stats, float invE) {
    int f = threadIdx.x;
    if (f < 32) {
        float mu = stats[f] * invE;
        float var = stats[32 + f] * invE - mu * mu;
        float rs = rsqrtf(var + EPS);
        float sc = rs * gamma[f];
        stats[64 + f] = sc;                // scale
        stats[96 + f] = beta[f] - mu * sc; // shift
    }
}

// Normalize edges to f16, accumulate x_init segment sums + counts at dst.
// 32 threads per edge (one per feature), 8 edges per block.
__global__ __launch_bounds__(256) void normalize_scatter_kernel(
    const float* __restrict__ edge_attr, const int* __restrict__ dst,
    const float* __restrict__ stats, _Float16* __restrict__ ea16,
    float* __restrict__ xinit_sum, float* __restrict__ count, int E) {
    const int f = threadIdx.x & 31;
    const int e = blockIdx.x * 8 + (threadIdx.x >> 5);
    if (e >= E) return;
    float v = edge_attr[(size_t)e * 32 + f] * stats[64 + f] + stats[96 + f];
    ea16[(size_t)e * 32 + f] = (_Float16)v;
    int d = dst[e];
    atomAddF(&xinit_sum[(size_t)d * 32 + f], v);
    if (f == 0) atomAddF(&count[d], 1.0f);
}

// ---------------------------------------------------------------------------
// Node init: x = relu((xinit_sum/count) @ W_init + b_init)  -> x16 [N,64]
// One wave per 16-node tile; K=32 -> 1 WMMA per N-tile.
// ---------------------------------------------------------------------------
__global__ __launch_bounds__(256) void node_init_kernel(
    const float* __restrict__ xinit_sum, const float* __restrict__ count,
    const _Float16* __restrict__ wt_init, const float* __restrict__ b_init,
    _Float16* __restrict__ x16, int N) {
    const int lane = threadIdx.x & 31;
    const int gwave = blockIdx.x * 8 + (threadIdx.x >> 5);
    const int tiles = (N + 15) / 16;
    if (gwave >= tiles) return;
    const int m = lane & 15, kq = lane >> 4, nl = lane & 15;
    const int node = min(gwave * 16 + m, N - 1);
    const float inv = 1.0f / fmaxf(count[node], 1.0f);
    v16h a = load_a_f32_scaled(xinit_sum + (size_t)node * 32, 0, kq, inv);
    v8f acc[4];
#pragma unroll
    for (int n = 0; n < 4; ++n) {
        v16h b = load_b_frag(wt_init, 32, n * 16, 0, lane);
        v8f c = splat8(b_init[n * 16 + nl]);
        acc[n] = wmma32(a, b, c);
    }
    const int row0 = gwave * 16 + 8 * kq;
    if (gwave * 16 + 16 <= N) {         // full tile: unguarded stores
#pragma unroll
        for (int n = 0; n < 4; ++n)
#pragma unroll
            for (int r = 0; r < 8; ++r)
                x16[(size_t)(row0 + r) * 64 + n * 16 + nl] = (_Float16)fmaxf(acc[n][r], 0.0f);
    } else {
#pragma unroll
        for (int n = 0; n < 4; ++n)
#pragma unroll
            for (int r = 0; r < 8; ++r)
                if (row0 + r < N)
                    x16[(size_t)(row0 + r) * 64 + n * 16 + nl] = (_Float16)fmaxf(acc[n][r], 0.0f);
    }
}

// ---------------------------------------------------------------------------
// Edge message kernel (the hot loop): per 16-edge tile,
//   m = relu([ea | x[src]] @ W_e1 + b1); m = relu(m @ W_e2 + b2);
//   atomic scatter-add m into agg[dst].
// B fragments (20 WMMA operands) live in registers across an 8-tile loop.
// ---------------------------------------------------------------------------
__global__ __launch_bounds__(256) void edge_layer_kernel(
    const _Float16* __restrict__ ea16, const _Float16* __restrict__ x16,
    const int* __restrict__ src, const int* __restrict__ dst,
    const _Float16* __restrict__ wt_e1, const float* __restrict__ b_e1,
    const _Float16* __restrict__ wt_e2, const float* __restrict__ b_e2,
    float* __restrict__ agg, int E, int tiles, int tpw) {
    __shared__ __align__(16) _Float16 smem[8][16 * 64];
    const int lane = threadIdx.x & 31;
    const int wib = threadIdx.x >> 5;
    const int gwave = blockIdx.x * 8 + wib;
    const int m = lane & 15, kq = lane >> 4, nl = lane & 15;
    _Float16* myLds = smem[wib];

    // Hoist weight B-fragments + bias splats
    v16h B1[3][4], B2[2][4];
    float b1s[4], b2s[4];
#pragma unroll
    for (int n = 0; n < 4; ++n) {
#pragma unroll
        for (int s = 0; s < 3; ++s) B1[s][n] = load_b_frag(wt_e1, 96, n * 16, s * 32, lane);
#pragma unroll
        for (int s = 0; s < 2; ++s) B2[s][n] = load_b_frag(wt_e2, 64, n * 16, s * 32, lane);
        b1s[n] = b_e1[n * 16 + nl];
        b2s[n] = b_e2[n * 16 + nl];
    }

    const int t0 = gwave * tpw;
    const int t1 = min(t0 + tpw, tiles);
    for (int t = t0; t < t1; ++t) {
        const int ebase = t * 16;
        // prefetch next tile's streaming data (global_prefetch_b8)
        if (t + 1 < t1) {
            const int en = min((t + 1) * 16 + m, E - 1);
            __builtin_prefetch(ea16 + (size_t)en * 32, 0, 0);
            __builtin_prefetch(src + en, 0, 0);
            __builtin_prefetch(dst + en, 0, 0);
        }
        const int e = min(ebase + m, E - 1);
        // A fragments: K 0..31 = ea, K 32..95 = x[src]
        v16h a0 = load_a_row(ea16 + (size_t)e * 32, 0, kq);
        const _Float16* xr = x16 + (size_t)src[e] * 64;
        v16h a1 = load_a_row(xr, 0, kq);
        v16h a2 = load_a_row(xr, 32, kq);

        // GEMM1: [16,96] x [96,64]
        v8f acc[4];
#pragma unroll
        for (int n = 0; n < 4; ++n) {
            v8f c = splat8(b1s[n]);
            c = wmma32(a0, B1[0][n], c);
            c = wmma32(a1, B1[1][n], c);
            c = wmma32(a2, B1[2][n], c);
            acc[n] = c;
        }
        // ReLU -> wave-private LDS tile (D-layout -> row-major for A reload)
#pragma unroll
        for (int n = 0; n < 4; ++n)
#pragma unroll
            for (int r = 0; r < 8; ++r)
                myLds[(r + 8 * kq) * 64 + n * 16 + nl] = (_Float16)fmaxf(acc[n][r], 0.0f);
        wave_ds_fence();
        v16h ma0 = load_a_row(myLds + m * 64, 0, kq);
        v16h ma1 = load_a_row(myLds + m * 64, 32, kq);

        // GEMM2: [16,64] x [64,64]
        v8f acc2[4];
#pragma unroll
        for (int n = 0; n < 4; ++n) {
            v8f c = splat8(b2s[n]);
            c = wmma32(ma0, B2[0][n], c);
            c = wmma32(ma1, B2[1][n], c);
            acc2[n] = c;
        }
        // ReLU + atomic segment-sum scatter
        const int er0 = ebase + 8 * kq;
        if (ebase + 16 <= E) {          // full tile: unguarded scatter
#pragma unroll
            for (int r = 0; r < 8; ++r) {
                const int d = dst[er0 + r];
                float* ag = agg + (size_t)d * 64 + nl;
#pragma unroll
                for (int n = 0; n < 4; ++n)
                    atomAddF(ag + n * 16, fmaxf(acc2[n][r], 0.0f));
            }
        } else {
#pragma unroll
            for (int r = 0; r < 8; ++r) {
                if (er0 + r < E) {
                    const int d = dst[er0 + r];
                    float* ag = agg + (size_t)d * 64 + nl;
#pragma unroll
                    for (int n = 0; n < 4; ++n)
                        atomAddF(ag + n * 16, fmaxf(acc2[n][r], 0.0f));
                }
            }
        }
    }
}

// ---------------------------------------------------------------------------
// Node update: x = relu((agg/count) @ W_u + b_u) -> x16   (K=64 -> 2 WMMA)
// ---------------------------------------------------------------------------
__global__ __launch_bounds__(256) void node_update_kernel(
    const float* __restrict__ agg, const float* __restrict__ count,
    const _Float16* __restrict__ wt_u, const float* __restrict__ b_u,
    _Float16* __restrict__ x16, int N) {
    const int lane = threadIdx.x & 31;
    const int gwave = blockIdx.x * 8 + (threadIdx.x >> 5);
    const int tiles = (N + 15) / 16;
    if (gwave >= tiles) return;
    const int m = lane & 15, kq = lane >> 4, nl = lane & 15;
    const int node = min(gwave * 16 + m, N - 1);
    const float inv = 1.0f / fmaxf(count[node], 1.0f);
    const float* row = agg + (size_t)node * 64;
    v16h a0 = load_a_f32_scaled(row, 0, kq, inv);
    v16h a1 = load_a_f32_scaled(row, 32, kq, inv);
    v8f acc[4];
#pragma unroll
    for (int n = 0; n < 4; ++n) {
        v16h bf0 = load_b_frag(wt_u, 64, n * 16, 0, lane);
        v16h bf1 = load_b_frag(wt_u, 64, n * 16, 32, lane);
        v8f c = splat8(b_u[n * 16 + nl]);
        c = wmma32(a0, bf0, c);
        acc[n] = wmma32(a1, bf1, c);
    }
    const int row0 = gwave * 16 + 8 * kq;
    if (gwave * 16 + 16 <= N) {
#pragma unroll
        for (int n = 0; n < 4; ++n)
#pragma unroll
            for (int r = 0; r < 8; ++r)
                x16[(size_t)(row0 + r) * 64 + n * 16 + nl] = (_Float16)fmaxf(acc[n][r], 0.0f);
    } else {
#pragma unroll
        for (int n = 0; n < 4; ++n)
#pragma unroll
            for (int r = 0; r < 8; ++r)
                if (row0 + r < N)
                    x16[(size_t)(row0 + r) * 64 + n * 16 + nl] = (_Float16)fmaxf(acc[n][r], 0.0f);
    }
}

// ---------------------------------------------------------------------------
// Output: out = x @ W_out + b_out  (f32, no relu)
// ---------------------------------------------------------------------------
__global__ __launch_bounds__(256) void node_out_kernel(
    const _Float16* __restrict__ x16, const _Float16* __restrict__ wt_out,
    const float* __restrict__ b_out, float* __restrict__ out, int N) {
    const int lane = threadIdx.x & 31;
    const int gwave = blockIdx.x * 8 + (threadIdx.x >> 5);
    const int tiles = (N + 15) / 16;
    if (gwave >= tiles) return;
    const int m = lane & 15, kq = lane >> 4, nl = lane & 15;
    const int node = min(gwave * 16 + m, N - 1);
    const _Float16* row = x16 + (size_t)node * 64;
    v16h a0 = load_a_row(row, 0, kq);
    v16h a1 = load_a_row(row, 32, kq);
    v8f acc[4];
#pragma unroll
    for (int n = 0; n < 4; ++n) {
        v16h bf0 = load_b_frag(wt_out, 64, n * 16, 0, lane);
        v16h bf1 = load_b_frag(wt_out, 64, n * 16, 32, lane);
        v8f c = splat8(b_out[n * 16 + nl]);
        c = wmma32(a0, bf0, c);
        acc[n] = wmma32(a1, bf1, c);
    }
    const int row0 = gwave * 16 + 8 * kq;
    if (gwave * 16 + 16 <= N) {
#pragma unroll
        for (int n = 0; n < 4; ++n)
#pragma unroll
            for (int r = 0; r < 8; ++r)
                out[(size_t)(row0 + r) * 64 + n * 16 + nl] = acc[n][r];
    } else {
#pragma unroll
        for (int n = 0; n < 4; ++n)
#pragma unroll
            for (int r = 0; r < 8; ++r)
                if (row0 + r < N)
                    out[(size_t)(row0 + r) * 64 + n * 16 + nl] = acc[n][r];
    }
}

// ---------------------------------------------------------------------------
// Host-side orchestration
// ---------------------------------------------------------------------------
extern "C" void kernel_launch(void* const* d_in, const int* in_sizes, int n_in,
                              void* d_out, int out_size, void* d_ws, size_t ws_size,
                              hipStream_t stream) {
    const int*   edge_index = (const int*)d_in[0];
    const float* edge_attr  = (const float*)d_in[1];
    const float* gamma      = (const float*)d_in[2];
    const float* beta       = (const float*)d_in[3];
    const float* W_init     = (const float*)d_in[4];
    const float* b_init     = (const float*)d_in[5];
    const float* W_e1       = (const float*)d_in[6];
    const float* b_e1       = (const float*)d_in[7];
    const float* W_e2       = (const float*)d_in[8];
    const float* b_e2       = (const float*)d_in[9];
    const float* W_u        = (const float*)d_in[10];
    const float* b_u        = (const float*)d_in[11];
    const float* W_out      = (const float*)d_in[12];
    const float* b_out      = (const float*)d_in[13];

    const int E = in_sizes[1] / 32;      // edge_attr is [E,32]
    const int N = out_size / 64;         // output is [N,64]
    const int* src = edge_index;
    const int* dst = edge_index + E;

    // ---- workspace layout (256B aligned) ----
    char* ws = (char*)d_ws;
    size_t o = 0;
    auto alloc = [&](size_t bytes) { void* p = ws + o; o += (bytes + 255) & ~(size_t)255; return p; };
    float*     stats     = (float*)alloc(256 * sizeof(float));        // sum/sumsq/scale/shift
    _Float16*  ea16      = (_Float16*)alloc((size_t)E * 32 * 2);
    _Float16*  x16       = (_Float16*)alloc((size_t)N * 64 * 2);
    float*     xinit_sum = (float*)alloc((size_t)N * 32 * 4);
    float*     count     = (float*)alloc((size_t)N * 4);
    float*     agg       = (float*)alloc((size_t)N * 64 * 4);
    _Float16*  wt_init   = (_Float16*)alloc(64 * 32 * 2);
    _Float16*  wt_e1     = (_Float16*)alloc(64 * 96 * 2);
    _Float16*  wt_e2     = (_Float16*)alloc(64 * 64 * 2);
    _Float16*  wt_u      = (_Float16*)alloc(64 * 64 * 2);
    _Float16*  wt_out    = (_Float16*)alloc(64 * 64 * 2);
    (void)ws_size; (void)n_in;

    auto zero = [&](float* p, long n) {
        int blocks = (int)((n + 255) / 256);
        if (blocks > 2048) blocks = 2048;
        zero_f32_kernel<<<blocks, 256, 0, stream>>>(p, n);
    };

    // 1) zero accumulators
    zero(stats, 256);
    zero(xinit_sum, (long)N * 32);
    zero(count, N);

    // 2) weight prep (transpose + f16)
    weight_prep_kernel<<<1, 256, 0, stream>>>(W_init, W_e1, W_e2, W_u, W_out,
                                              wt_init, wt_e1, wt_e2, wt_u, wt_out);

    // 3) BatchNorm statistics
    {
        const int blocks = 1024;
        const int chunk = (E + blocks - 1) / blocks;
        bn_partial_kernel<<<blocks, 256, 0, stream>>>(edge_attr, stats, E, chunk);
        bn_finalize_kernel<<<1, 32, 0, stream>>>(gamma, beta, stats, 1.0f / (float)E);
    }

    // 4) normalize edges + x_init segment sums + counts
    normalize_scatter_kernel<<<(E + 7) / 8, 256, 0, stream>>>(
        edge_attr, dst, stats, ea16, xinit_sum, count, E);

    // 5) node init GEMM
    const int ntiles = (N + 15) / 16;
    const int nblocks = (ntiles + 7) / 8;
    node_init_kernel<<<nblocks, 256, 0, stream>>>(xinit_sum, count, wt_init, b_init, x16, N);

    // 6) three weight-shared message-passing layers
    const int etiles = (E + 15) / 16;
    const int tpw = 8;                              // tiles per wave (amortize B-frag loads)
    const int ewaves = (etiles + tpw - 1) / tpw;
    const int eblocks = (ewaves + 7) / 8;
    for (int layer = 0; layer < 3; ++layer) {
        zero(agg, (long)N * 64);
        edge_layer_kernel<<<eblocks, 256, 0, stream>>>(
            ea16, x16, src, dst, wt_e1, b_e1, wt_e2, b_e2, agg, E, etiles, tpw);
        node_update_kernel<<<nblocks, 256, 0, stream>>>(agg, count, wt_u, b_u, x16, N);
    }

    // 7) output projection
    node_out_kernel<<<nblocks, 256, 0, stream>>>(x16, wt_out, b_out, (float*)d_out, N);
}
